// MultiHeadSelfAttention_62139586838556
// MI455X (gfx1250) — compile-verified, matrix operand order checked by reading, not run on hardware
//
#include <hip/hip_runtime.h>

typedef __bf16 bf16;
typedef __attribute__((ext_vector_type(16))) __bf16 v16bf;
typedef __attribute__((ext_vector_type(8)))  __bf16 v8bf;
typedef __attribute__((ext_vector_type(8)))  float  v8f;
typedef __attribute__((ext_vector_type(4)))  unsigned v4u;
typedef __attribute__((ext_vector_type(4)))  int v4i;
typedef __attribute__((ext_vector_type(8)))  int v8i;

// Problem constants
constexpr int BATCH = 4;
constexpr int SEQ   = 2048;
constexpr int DIM   = 1024;
constexpr int HEADS = 16;
constexpr int DK    = 64;
constexpr int MROWS = BATCH * SEQ;           // 8192

// LDS row strides for attention (padded to dodge bank conflicts)
constexpr int SROW = 2064;                   // floats per score row
constexpr int PROW = 2080;                   // bf16 per prob row
constexpr int SMEM_ATTN = 16 * SROW * 4 + 16 * PROW * 2;   // 198,656 B < 320 KB

// Tensor Data Mover availability (5-arg builtin on ROCm7.2/clang-22,
// 6-arg on amdgpu-toolchain/clang-23)
#if defined(__has_builtin)
#  if __has_builtin(__builtin_amdgcn_tensor_load_to_lds) && \
      __has_builtin(__builtin_amdgcn_s_wait_tensorcnt)
#    define HAVE_TDM 1
#  endif
#endif
#ifndef HAVE_TDM
#  define HAVE_TDM 0
#endif

// ---------------------------------------------------------------------------
// WMMA helpers: CDNA5 bf16 16x16x32 fragment layout.
//   row/col = lane & 15;  half j -> K = kk0 + 16*(j>>3) + 8*(lane>>4) + (j&7)
// => two contiguous 8-half (16 B) chunks at +0 and +16 halves.
// ---------------------------------------------------------------------------
__device__ __forceinline__ v16bf load_frag_bf16(const bf16* row_ptr, int kk0, int lane) {
    const bf16* p = row_ptr + kk0 + ((lane >> 4) << 3);
    union { v16bf v; v8bf h[2]; } u;
    u.h[0] = *(const v8bf*)(p);
    u.h[1] = *(const v8bf*)(p + 16);
    return u.v;
}

__device__ __forceinline__ v8f wmma_bf16(v16bf a, v16bf b, v8f c) {
    return __builtin_amdgcn_wmma_f32_16x16x32_bf16(false, a, false, b,
                                                   (short)0, c, false, false);
}

#if HAVE_TDM
// ---------------------------------------------------------------------------
// TDM: 2D bf16 tile (tile_x contiguous elems per row, tile_y rows, row stride
// in elements) DMA'd from global to a contiguous LDS region.
// D# per CDNA5 ISA.
// ---------------------------------------------------------------------------
__device__ __forceinline__ void tdm_load_2d(unsigned lds_byte_addr, const void* gaddr,
                                            unsigned tile_x, unsigned tile_y,
                                            unsigned tensor_x, unsigned tensor_y,
                                            unsigned row_stride_elems) {
    unsigned long long ga = (unsigned long long)(size_t)gaddr;
    v4u g0;
    g0.x = 1u;                                              // count=1 (valid user D#)
    g0.y = lds_byte_addr;                                   // [63:32] lds_addr
    g0.z = (unsigned)ga;                                    // [95:64] global_addr lo
    g0.w = (unsigned)((ga >> 32) & 0x01FFFFFFu) | (2u << 30); // ga hi + type=2
    v8i g1;
    g1[0] = (int)(1u << 16);                                // data_size=1 -> 2 bytes
    g1[1] = (int)((tensor_x & 0xFFFFu) << 16);              // tensor_dim0[15:0]
    g1[2] = (int)((tensor_x >> 16) | ((tensor_y & 0xFFFFu) << 16));
    g1[3] = (int)((tensor_y >> 16) | (tile_x << 16));       // tile_dim0
    g1[4] = (int)tile_y;                                    // tile_dim1 (tile_dim2=0)
    g1[5] = (int)row_stride_elems;                          // tensor_dim0_stride lo
    g1[6] = 0;
    g1[7] = 0;
    v4i z4 = {0, 0, 0, 0};
#if defined(__clang_major__) && __clang_major__ >= 23
    v8i z8 = {};
    __builtin_amdgcn_tensor_load_to_lds(g0, g1, z4, z4, z8, 0);
#else
    __builtin_amdgcn_tensor_load_to_lds(g0, g1, z4, z4, 0);
#endif
}
#endif

// ---------------------------------------------------------------------------
// fp32 -> bf16 elementwise convert
// ---------------------------------------------------------------------------
__global__ void convert_f32_bf16(const float* __restrict__ in,
                                 bf16* __restrict__ out, int n) {
    int i = blockIdx.x * blockDim.x + threadIdx.x;
    if (i < n) out[i] = (bf16)in[i];
}

// ---------------------------------------------------------------------------
// 1024x1024 transpose + convert: out[n][k] = (bf16) in[k][n]
// ---------------------------------------------------------------------------
__global__ void transpose_w(const float* __restrict__ in, bf16* __restrict__ out) {
    __shared__ float tile[32][33];
    int bx = blockIdx.x * 32, by = blockIdx.y * 32;
    for (int i = threadIdx.y; i < 32; i += 8)
        tile[i][threadIdx.x] = in[(size_t)(by + i) * DIM + bx + threadIdx.x];
    __syncthreads();
    for (int i = threadIdx.y; i < 32; i += 8)
        out[(size_t)(bx + i) * DIM + by + threadIdx.x] = (bf16)tile[threadIdx.x][i];
}

// ---------------------------------------------------------------------------
// QKV projection: xb[8192,1024] @ W^T (+bias).
// Block = 8 waves * 16 rows = 128 rows x 64 cols; B tile (64x32 bf16 = 4KB)
// staged in LDS via TDM double-buffer; each wave: 1 A-frag -> 4 WMMAs.
// mode (blockIdx.z): 0->Q (scaled, [B,H,S,64]) 1->K ([B,H,S,64]) 2->V ([B,H,64,S])
// ---------------------------------------------------------------------------
__global__ void qkv_proj(const bf16* __restrict__ xb,
                         const bf16* __restrict__ WqT, const bf16* __restrict__ WkT,
                         const bf16* __restrict__ WvT,
                         const float* __restrict__ bq, const float* __restrict__ bk,
                         const float* __restrict__ bv,
                         bf16* __restrict__ Qb, bf16* __restrict__ Kb,
                         bf16* __restrict__ Vt) {
    __shared__ alignas(128) bf16 ldsB[2][64 * 32];
    const int lane = threadIdx.x & 31;
    const int wave = threadIdx.x >> 5;
    const int mode = blockIdx.z;
    const bf16*  Wt   = (mode == 0) ? WqT : (mode == 1) ? WkT : WvT;
    const float* bias = (mode == 0) ? bq  : (mode == 1) ? bk  : bv;

    const int n0 = blockIdx.x * 64;
    const int m0 = (blockIdx.y * 8 + wave) * 16;
    const bf16* arow  = xb + (size_t)(m0 + (lane & 15)) * DIM;
    const bf16* btile = Wt + (size_t)n0 * DIM;      // 64 rows x DIM, row-major

    v8f acc[4] = {{}, {}, {}, {}};
#if HAVE_TDM
    if (wave == 0)
        tdm_load_2d((unsigned)(size_t)&ldsB[0][0], btile, 32, 64, DIM, DIM, DIM);
#endif
    for (int k = 0; k < DIM / 32; ++k) {
        int buf = k & 1;
#if HAVE_TDM
        if (wave == 0) __builtin_amdgcn_s_wait_tensorcnt(0);
        __syncthreads();                       // buf[k&1] ready, prior reads retired
        if (wave == 0 && k + 1 < DIM / 32)
            tdm_load_2d((unsigned)(size_t)&ldsB[buf ^ 1][0], btile + (size_t)(k + 1) * 32,
                        32, 64, DIM, DIM, DIM);
#else
        buf = 0;
        {   // cooperative copy fallback: 256 threads x 16 B = 4 KB tile
            int t = threadIdx.x;
            const v8bf* src = (const v8bf*)(btile + (size_t)(t >> 2) * DIM +
                                            (size_t)k * 32 + (t & 3) * 8);
            *(v8bf*)&ldsB[0][(t >> 2) * 32 + (t & 3) * 8] = *src;
        }
        __syncthreads();
#endif
        v16bf a = load_frag_bf16(arow, k * 32, lane);
        for (int nt = 0; nt < 4; ++nt) {
            v16bf b = load_frag_bf16(&ldsB[buf][(nt * 16 + (lane & 15)) * 32], 0, lane);
            acc[nt] = wmma_bf16(a, b, acc[nt]);
        }
#if !HAVE_TDM
        __syncthreads();
#endif
    }

    for (int nt = 0; nt < 4; ++nt) {
        int n = n0 + nt * 16 + (lane & 15);
        float bi = bias[n];
        int h = n >> 6, d = n & 63;
        for (int r = 0; r < 8; ++r) {
            int m = m0 + r + ((lane >> 4) << 3);
            int bm = m >> 11, s = m & (SEQ - 1);
            size_t bh = (size_t)bm * HEADS + h;
            float v = acc[nt][r] + bi;
            if (mode == 0)      Qb[(bh * SEQ + s) * DK + d] = (bf16)(v * 0.125f);
            else if (mode == 1) Kb[(bh * SEQ + s) * DK + d] = (bf16)v;
            else                Vt[(bh * DK + d) * SEQ + s] = (bf16)v;
        }
    }
}

// ---------------------------------------------------------------------------
// Fused attention per (b,h,16-query tile): scores -> LDS, softmax,
// normalized weights -> HBM, P@V -> attn buffer.  128 threads = 4 waves.
// ---------------------------------------------------------------------------
__global__ void attention_fused(const bf16* __restrict__ Qb, const bf16* __restrict__ Kb,
                                const bf16* __restrict__ Vt,
                                bf16* __restrict__ attnb, float* __restrict__ wout) {
    extern __shared__ char smem[];
    float* Ssc = (float*)smem;
    bf16*  Pb  = (bf16*)(smem + 16 * SROW * sizeof(float));

    const int lane = threadIdx.x & 31;
    const int wave = threadIdx.x >> 5;
    const int blk  = blockIdx.x;
    const int qt = blk & 127, bh = blk >> 7;
    const int q0 = qt * 16;

    const bf16* Qh = Qb + (size_t)bh * SEQ * DK;
    const bf16* Kh = Kb + (size_t)bh * SEQ * DK;
    const bf16* Vh = Vt + (size_t)bh * DK * SEQ;

    // ---- Phase 1: scores = Q K^T (pre-scaled Q); 16 x 2048 into LDS ----
    const bf16* qrow = Qh + (size_t)(q0 + (lane & 15)) * DK;
    v16bf a0 = load_frag_bf16(qrow, 0, lane);
    v16bf a1 = load_frag_bf16(qrow, 32, lane);
    for (int kt = wave; kt < SEQ / 16; kt += 4) {
        const bf16* krow = Kh + (size_t)(kt * 16 + (lane & 15)) * DK;
        v16bf b0 = load_frag_bf16(krow, 0, lane);
        v16bf b1 = load_frag_bf16(krow, 32, lane);
        v8f c = {};
        c = wmma_bf16(a0, b0, c);
        c = wmma_bf16(a1, b1, c);
        int col = kt * 16 + (lane & 15);
        for (int r = 0; r < 8; ++r)
            Ssc[(r + ((lane >> 4) << 3)) * SROW + col] = c[r];
    }
    __syncthreads();

    // ---- Phase 2: row softmax; normalized weights -> HBM + bf16 LDS ----
    for (int row = wave * 4; row < wave * 4 + 4; ++row) {
        float* srow = Ssc + row * SROW;
        float mx = -1e30f;
        for (int c2 = lane; c2 < SEQ; c2 += 32) mx = fmaxf(mx, srow[c2]);
        for (int off = 16; off; off >>= 1) mx = fmaxf(mx, __shfl_xor(mx, off, 32));
        float sum = 0.f;
        for (int c2 = lane; c2 < SEQ; c2 += 32) {
            float e = __expf(srow[c2] - mx);
            srow[c2] = e;
            sum += e;
        }
        for (int off = 16; off; off >>= 1) sum += __shfl_xor(sum, off, 32);
        float inv = 1.f / sum;
        float* wrow = wout + ((size_t)bh * SEQ + q0 + row) * SEQ;
        bf16*  prow = Pb + row * PROW;
        for (int c2 = lane; c2 < SEQ; c2 += 32) {
            float p = srow[c2] * inv;
            prow[c2] = (bf16)p;
            wrow[c2] = p;
        }
    }
    __syncthreads();

    // ---- Phase 3: attn = P @ V ; wave w owns dv columns [16w, 16w+16) ----
    const int n0 = wave * 16;
    const bf16* prow0 = Pb + (size_t)(lane & 15) * PROW;
    const bf16* vrow  = Vh + (size_t)(n0 + (lane & 15)) * SEQ;
    v8f c = {};
    for (int kk = 0; kk < SEQ; kk += 32) {
        v16bf a = load_frag_bf16(prow0, kk, lane);
        v16bf b = load_frag_bf16(vrow, kk, lane);
        c = wmma_bf16(a, b, c);
    }
    const int h = bh & (HEADS - 1), bidx = bh >> 4;
    const int coln = h * DK + n0 + (lane & 15);
    for (int r = 0; r < 8; ++r) {
        int s = q0 + r + ((lane >> 4) << 3);
        attnb[((size_t)bidx * SEQ + s) * DIM + coln] = (bf16)c[r];
    }
}

// ---------------------------------------------------------------------------
// Output projection: attn[8192,1024] @ Wo (+bo) -> out fp32.
// Same TDM-staged block structure as qkv_proj.
// ---------------------------------------------------------------------------
__global__ void out_proj(const bf16* __restrict__ attnb, const bf16* __restrict__ WoT,
                         const float* __restrict__ bo, float* __restrict__ out) {
    __shared__ alignas(128) bf16 ldsB[2][64 * 32];
    const int lane = threadIdx.x & 31;
    const int wave = threadIdx.x >> 5;
    const int n0 = blockIdx.x * 64;
    const int m0 = (blockIdx.y * 8 + wave) * 16;
    const bf16* arow  = attnb + (size_t)(m0 + (lane & 15)) * DIM;
    const bf16* btile = WoT + (size_t)n0 * DIM;

    v8f acc[4] = {{}, {}, {}, {}};
#if HAVE_TDM
    if (wave == 0)
        tdm_load_2d((unsigned)(size_t)&ldsB[0][0], btile, 32, 64, DIM, DIM, DIM);
#endif
    for (int k = 0; k < DIM / 32; ++k) {
        int buf = k & 1;
#if HAVE_TDM
        if (wave == 0) __builtin_amdgcn_s_wait_tensorcnt(0);
        __syncthreads();
        if (wave == 0 && k + 1 < DIM / 32)
            tdm_load_2d((unsigned)(size_t)&ldsB[buf ^ 1][0], btile + (size_t)(k + 1) * 32,
                        32, 64, DIM, DIM, DIM);
#else
        buf = 0;
        {
            int t = threadIdx.x;
            const v8bf* src = (const v8bf*)(btile + (size_t)(t >> 2) * DIM +
                                            (size_t)k * 32 + (t & 3) * 8);
            *(v8bf*)&ldsB[0][(t >> 2) * 32 + (t & 3) * 8] = *src;
        }
        __syncthreads();
#endif
        v16bf a = load_frag_bf16(arow, k * 32, lane);
        for (int nt = 0; nt < 4; ++nt) {
            v16bf b = load_frag_bf16(&ldsB[buf][(nt * 16 + (lane & 15)) * 32], 0, lane);
            acc[nt] = wmma_bf16(a, b, acc[nt]);
        }
#if !HAVE_TDM
        __syncthreads();
#endif
    }

    for (int nt = 0; nt < 4; ++nt) {
        int n = n0 + nt * 16 + (lane & 15);
        float bi = bo[n];
        for (int r = 0; r < 8; ++r) {
            int m = m0 + r + ((lane >> 4) << 3);
            out[(size_t)m * DIM + n] = acc[nt][r] + bi;
        }
    }
}

// ---------------------------------------------------------------------------
extern "C" void kernel_launch(void* const* d_in, const int* in_sizes, int n_in,
                              void* d_out, int out_size, void* d_ws, size_t ws_size,
                              hipStream_t stream) {
    const float* x  = (const float*)d_in[0];
    const float* Wq = (const float*)d_in[1];
    const float* bq = (const float*)d_in[2];
    const float* Wk = (const float*)d_in[3];
    const float* bk = (const float*)d_in[4];
    const float* Wv = (const float*)d_in[5];
    const float* bv = (const float*)d_in[6];
    const float* Wo = (const float*)d_in[7];
    const float* bo = (const float*)d_in[8];

    float* out     = (float*)d_out;                       // [B,S,D]
    float* weights = out + (size_t)MROWS * DIM;           // [B,H,S,S]

    // Workspace carve-up (bf16 everywhere), ~88 MB total
    char* w = (char*)d_ws;
    bf16* xb   = (bf16*)w; w += (size_t)MROWS * DIM * 2;
    bf16* WqT  = (bf16*)w; w += (size_t)DIM * DIM * 2;
    bf16* WkT  = (bf16*)w; w += (size_t)DIM * DIM * 2;
    bf16* WvT  = (bf16*)w; w += (size_t)DIM * DIM * 2;
    bf16* WoT  = (bf16*)w; w += (size_t)DIM * DIM * 2;
    bf16* Qb   = (bf16*)w; w += (size_t)MROWS * DIM * 2;  // [B,H,S,64]
    bf16* Kb   = (bf16*)w; w += (size_t)MROWS * DIM * 2;  // [B,H,S,64]
    bf16* Vt   = (bf16*)w; w += (size_t)MROWS * DIM * 2;  // [B,H,64,S]
    bf16* attn = (bf16*)w; w += (size_t)MROWS * DIM * 2;  // [B,S,H*dv]

    // 1) Convert x and transpose/convert the four weight matrices
    convert_f32_bf16<<<(MROWS * DIM + 255) / 256, 256, 0, stream>>>(x, xb, MROWS * DIM);
    dim3 tgrid(DIM / 32, DIM / 32), tblk(32, 8);
    transpose_w<<<tgrid, tblk, 0, stream>>>(Wq, WqT);
    transpose_w<<<tgrid, tblk, 0, stream>>>(Wk, WkT);
    transpose_w<<<tgrid, tblk, 0, stream>>>(Wv, WvT);
    transpose_w<<<tgrid, tblk, 0, stream>>>(Wo, WoT);

    // 2) QKV projections: block = 128 rows x 64 cols, z selects Q/K/V
    qkv_proj<<<dim3(DIM / 64, MROWS / 128, 3), 256, 0, stream>>>(
        xb, WqT, WkT, WvT, bq, bk, bv, Qb, Kb, Vt);

    // 3) Fused attention: one block per (b,h,16-query tile)
    attention_fused<<<dim3(BATCH * HEADS * (SEQ / 16)), 128, SMEM_ATTN, stream>>>(
        Qb, Kb, Vt, attn, weights);

    // 4) Output projection
    out_proj<<<dim3(DIM / 64, MROWS / 128), 256, 0, stream>>>(attn, WoT, bo, out);
}
